// StackedGRUInference_64699387347356
// MI455X (gfx1250) — compile-verified, matrix operand order checked by reading, not run on hardware
//
#include <hip/hip_runtime.h>

#define BB 4096
#define TT 512
#define FF 8
#define HH 64
#define CC 24
#define GG 192   // 3*H gate width

typedef __bf16 bf16;
typedef bf16 v16bf __attribute__((ext_vector_type(16)));
typedef float v8f  __attribute__((ext_vector_type(8)));
typedef float v2f  __attribute__((ext_vector_type(2)));

__device__ __forceinline__ v8f wmma_bf16(v16bf a, v16bf b, v8f c) {
  // D = A(16x32 bf16) x B(32x16 bf16) + C(16x16 f32)
  return __builtin_amdgcn_wmma_f32_16x16x32_bf16(false, a, false, b, (short)0, c, false, false);
}
__device__ __forceinline__ v8f wmma_f32(v2f a, v2f b, v8f c) {
  // D = A(16x4 f32) x B(4x16 f32) + C(16x16 f32)
  return __builtin_amdgcn_wmma_f32_16x16x4_f32(false, a, false, b, (short)0, c, false, false);
}
__device__ __forceinline__ v8f splat8(float v) { v8f r = {v, v, v, v, v, v, v, v}; return r; }

// Fast transcendental paths: raw v_rcp_f32 / v_exp_f32 / v_tanh_f32 (gfx1250 TRANS ops).
__device__ __forceinline__ float fast_rcp(float x) { return __builtin_amdgcn_rcpf(x); }
__device__ __forceinline__ float sigm(float x) { return fast_rcp(1.0f + __expf(-x)); }
__device__ __forceinline__ float tanh_f(float x) {
#if __has_builtin(__builtin_amdgcn_tanhf)
  return __builtin_amdgcn_tanhf(x);
#else
  return 1.0f - 2.0f * fast_rcp(1.0f + __expf(2.0f * x));
#endif
}

// 16-lane x 16-bf16 fragment (32 contiguous bytes per lane) from LDS.
// A 16x32: lane L -> row = L&15, K base = kslab*32 + (L>>4)*16
// B 32x16: lane L -> col = L&15 (= weight row g), same K packing.
__device__ __forceinline__ v16bf ldfrag(const bf16* base, int row, int kb, int ld) {
  return *(const v16bf*)(base + row * ld + kb);
}
// Build a bf16 B-fragment directly from a global f32 weight row (one-time init).
__device__ __forceinline__ v16bf mkfrag_global(const float* __restrict__ w, int row, int kb, int ld) {
  const float* wr = w + row * ld + kb;
  v16bf r;
  #pragma unroll
  for (int k = 0; k < 16; ++k) r[k] = (bf16)wr[k];
  return r;
}

__global__ void __launch_bounds__(128) gru_fused_kernel(
    const float* __restrict__ x,
    const float* __restrict__ w_ih0, const float* __restrict__ w_hh0,
    const float* __restrict__ b_ih0, const float* __restrict__ b_hh0,
    const float* __restrict__ w_ih1, const float* __restrict__ w_hh1,
    const float* __restrict__ b_ih1, const float* __restrict__ b_hh1,
    const float* __restrict__ w_fc1, const float* __restrict__ b_fc1,
    const float* __restrict__ w_fc2, const float* __restrict__ b_fc2,
    float* __restrict__ out)
{
  extern __shared__ char smem[];
  bf16* s_whh0 = (bf16*)smem;               // 192x64
  bf16* s_wih1 = s_whh0 + GG * HH;          // 192x64
  bf16* s_wfc1 = s_wih1 + GG * HH;          // 128x64
  bf16* s_wfc2 = s_wfc1 + 128 * 64;         // 32x128 (rows >= 24 zero-padded)
  bf16* s_h0   = s_wfc2 + 32 * 128;         // 2 x (16x64) ping-pong
  bf16* s_h1   = s_h0 + 2 * 16 * 64;        // 2 x (16x64) ping-pong
  bf16* s_hid  = s_h1 + 2 * 16 * 64;        // 16x128 head hidden
  float* s_x   = (float*)(s_hid + 16 * 128); // 2 x (16x8) fp32 x-tile ping-pong

  const int tid  = threadIdx.x;
  const int lane = tid & 31;
  const int wv   = tid >> 5;        // wave id 0..3 (splits gate-N dimension)
  const int n15  = lane & 15;
  const int hi   = lane >> 4;
  const int b0   = blockIdx.x * 16; // batch tile base row

  // ---------------- preload weights (f32 -> bf16) into LDS ----------------
  for (int i = tid; i < GG * HH; i += 128) {
    s_whh0[i] = (bf16)w_hh0[i];
    s_wih1[i] = (bf16)w_ih1[i];
  }
  for (int i = tid; i < 128 * 64; i += 128) s_wfc1[i] = (bf16)w_fc1[i];
  for (int i = tid; i < 32 * 128; i += 128) s_wfc2[i] = (i < CC * 128) ? (bf16)w_fc2[i] : (bf16)0.0f;
  for (int i = tid; i < 2 * 16 * 64; i += 128) { s_h0[i] = (bf16)0.0f; s_h1[i] = (bf16)0.0f; }

  const int xrow = tid >> 3, xf = tid & 7;  // one x element per thread per step
  s_x[xrow * 8 + xf] = x[((size_t)(b0 + xrow) * TT + 0) * FF + xf];

  // ---------------- per-lane constants (D-layout local) ----------------
  // Wave wv owns gate tiles {wv, wv+4, wv+8} -> gate rows g0, 64+g0, 128+g0,
  // all feeding hidden column g0, so the GRU pointwise update is lane-local.
  const int g0 = wv * 16 + n15;             // hidden column 0..63
  // Merged biases: r/z gates fold b_ih+b_hh into one accumulator init;
  // the n gate keeps xn (b_ih) and hn (b_hh) separate for tanh(xn + r*hn).
  const float bs0r = b_ih0[g0]        + b_hh0[g0];
  const float bs0z = b_ih0[64 + g0]   + b_hh0[64 + g0];
  const float b0xn = b_ih0[128 + g0];
  const float b0hn = b_hh0[128 + g0];
  const float bs1r = b_ih1[g0]        + b_hh1[g0];
  const float bs1z = b_ih1[64 + g0]   + b_hh1[64 + g0];
  const float b1xn = b_ih1[128 + g0];
  const float b1hn = b_hh1[128 + g0];

  // w_ih0 B-frags for the f32 16x16x4 WMMA (12 VGPRs, register-resident)
  v2f Bx0[3][2];
  #pragma unroll
  for (int jj = 0; jj < 3; ++jj) {
    const float* wr = w_ih0 + (jj * 64 + g0) * FF;
    Bx0[jj][0] = v2f{wr[hi * 2],     wr[hi * 2 + 1]};      // K 0..3
    Bx0[jj][1] = v2f{wr[4 + hi * 2], wr[4 + hi * 2 + 1]};  // K 4..7
  }

  const int kb0 = hi * 16;        // bf16 K base, slab 0
  const int kb1 = 32 + hi * 16;   // bf16 K base, slab 1

  // Whh1 B-frags register-resident (48 VGPRs), built straight from global f32
  // (no LDS staging needed; one-time cvt). Whh0/Wih1 frags are re-read from
  // LDS every step, which the compiler cannot hoist across barriers but which
  // overlap the mandatory h-state A-fragment loads in each phase.
  v16bf Bh1[3][2];
  #pragma unroll
  for (int jj = 0; jj < 3; ++jj) {
    const int g = jj * 64 + g0;
    Bh1[jj][0] = mkfrag_global(w_hh1, g, kb0, 64);
    Bh1[jj][1] = mkfrag_global(w_hh1, g, kb1, 64);
  }

  float h0r[8], h1r[8];
  #pragma unroll
  for (int e = 0; e < 8; ++e) { h0r[e] = 0.f; h1r[e] = 0.f; }

  __syncthreads();

  // ---------------- fused two-layer GRU scan over T ----------------
  for (int t = 0; t < TT; ++t) {
    const int p = t & 1;

    // prefetch x[t+1] into a register; hint-prefetch x[t+2]
    float xnext = 0.f;
    if (t + 1 < TT) {
      xnext = x[((size_t)(b0 + xrow) * TT + (t + 1)) * FF + xf];
      if (t + 2 < TT) __builtin_prefetch(&x[((size_t)(b0 + xrow) * TT + (t + 2)) * FF + xf], 0, 0);
    }

    // ---- phase 1: layer 0 ----
    const bf16* h0p = s_h0 + p * (16 * 64);
    v16bf a0 = ldfrag(h0p, n15, kb0, 64);
    v16bf a1 = ldfrag(h0p, n15, kb1, 64);
    const float* xt = s_x + p * (16 * 8);
    v2f xa0 = *(const v2f*)(xt + n15 * 8 + hi * 2);
    v2f xa1 = *(const v2f*)(xt + n15 * 8 + 4 + hi * 2);

    // r-gate: x-proj (f32 WMMA) + h-proj (bf16 WMMA) share one accumulator
    v8f ar = splat8(bs0r);
    ar = wmma_f32(xa0, Bx0[0][0], ar);
    ar = wmma_f32(xa1, Bx0[0][1], ar);
    ar = wmma_bf16(a0, ldfrag(s_whh0, g0, kb0, 64), ar);
    ar = wmma_bf16(a1, ldfrag(s_whh0, g0, kb1, 64), ar);
    // z-gate
    v8f az = splat8(bs0z);
    az = wmma_f32(xa0, Bx0[1][0], az);
    az = wmma_f32(xa1, Bx0[1][1], az);
    az = wmma_bf16(a0, ldfrag(s_whh0, 64 + g0, kb0, 64), az);
    az = wmma_bf16(a1, ldfrag(s_whh0, 64 + g0, kb1, 64), az);
    // n-gate: xn and hn separate
    v8f axn = splat8(b0xn);
    axn = wmma_f32(xa0, Bx0[2][0], axn);
    axn = wmma_f32(xa1, Bx0[2][1], axn);
    v8f ahn = splat8(b0hn);
    ahn = wmma_bf16(a0, ldfrag(s_whh0, 128 + g0, kb0, 64), ahn);
    ahn = wmma_bf16(a1, ldfrag(s_whh0, 128 + g0, kb1, 64), ahn);

    bf16* h0n = s_h0 + (1 - p) * (16 * 64);
    #pragma unroll
    for (int e = 0; e < 8; ++e) {
      const int m = e + hi * 8;
      const float r  = sigm(ar[e]);
      const float z  = sigm(az[e]);
      const float nn = tanh_f(axn[e] + r * ahn[e]);
      const float h  = (1.f - z) * nn + z * h0r[e];
      h0r[e] = h;
      h0n[m * 64 + g0] = (bf16)h;       // publish h0(t+1) (own column only)
    }
    if (t + 1 < TT) s_x[(1 - p) * (16 * 8) + xrow * 8 + xf] = xnext;

    __syncthreads();                     // h0(t+1) complete across waves

    // ---- phase 2: layer 1 ----
    v16bf d0 = ldfrag(h0n, n15, kb0, 64);       // A: h0(t+1)
    v16bf d1 = ldfrag(h0n, n15, kb1, 64);
    const bf16* h1p = s_h1 + p * (16 * 64);
    v16bf c0 = ldfrag(h1p, n15, kb0, 64);       // A: h1(t) (valid all iteration)
    v16bf c1 = ldfrag(h1p, n15, kb1, 64);

    v8f br = splat8(bs1r);
    br = wmma_bf16(d0, ldfrag(s_wih1, g0, kb0, 64), br);
    br = wmma_bf16(d1, ldfrag(s_wih1, g0, kb1, 64), br);
    br = wmma_bf16(c0, Bh1[0][0], br);
    br = wmma_bf16(c1, Bh1[0][1], br);
    v8f bz = splat8(bs1z);
    bz = wmma_bf16(d0, ldfrag(s_wih1, 64 + g0, kb0, 64), bz);
    bz = wmma_bf16(d1, ldfrag(s_wih1, 64 + g0, kb1, 64), bz);
    bz = wmma_bf16(c0, Bh1[1][0], bz);
    bz = wmma_bf16(c1, Bh1[1][1], bz);
    v8f bxn = splat8(b1xn);
    bxn = wmma_bf16(d0, ldfrag(s_wih1, 128 + g0, kb0, 64), bxn);
    bxn = wmma_bf16(d1, ldfrag(s_wih1, 128 + g0, kb1, 64), bxn);
    v8f bhn = splat8(b1hn);
    bhn = wmma_bf16(c0, Bh1[2][0], bhn);
    bhn = wmma_bf16(c1, Bh1[2][1], bhn);

    bf16* h1n = s_h1 + (1 - p) * (16 * 64);
    #pragma unroll
    for (int e = 0; e < 8; ++e) {
      const int m = e + hi * 8;
      const float r  = sigm(br[e]);
      const float z  = sigm(bz[e]);
      const float nn = tanh_f(bxn[e] + r * bhn[e]);
      const float h  = (1.f - z) * nn + z * h1r[e];
      h1r[e] = h;
      h1n[m * 64 + g0] = (bf16)h;       // publish h1(t+1)
    }
    __syncthreads();                     // h1(t+1) + next x tile complete
  }

  // ---------------- MLP head: tanh(h @ Wfc1^T + b1) @ Wfc2^T + b2 ----------------
  const bf16* hfin = s_h1 + (TT & 1) * (16 * 64);
  v16bf ha0 = ldfrag(hfin, n15, kb0, 64);
  v16bf ha1 = ldfrag(hfin, n15, kb1, 64);
  #pragma unroll
  for (int q = 0; q < 2; ++q) {
    const int tile = wv + 4 * q;
    const int g = tile * 16 + n15;       // hidden column 0..127
    v8f acc = splat8(b_fc1[g]);
    acc = wmma_bf16(ha0, ldfrag(s_wfc1, g, hi * 16, 64), acc);
    acc = wmma_bf16(ha1, ldfrag(s_wfc1, g, 32 + hi * 16, 64), acc);
    #pragma unroll
    for (int e = 0; e < 8; ++e) {
      const int m = e + hi * 8;
      s_hid[m * 128 + g] = (bf16)tanh_f(acc[e]);
    }
  }
  __syncthreads();

  if (wv < 2) {                          // wave-uniform branch; EXEC all-1 inside
    const int g = wv * 16 + n15;         // logit column 0..31 (valid < 24)
    v8f acc = splat8((g < CC) ? b_fc2[g] : 0.f);
    #pragma unroll
    for (int s = 0; s < 4; ++s) {        // K = 128 -> 4 slabs
      const int kb = s * 32 + hi * 16;
      v16bf aa = ldfrag(s_hid,  n15, kb, 128);
      v16bf bb = ldfrag(s_wfc2, g,   kb, 128);
      acc = wmma_bf16(aa, bb, acc);
    }
    if (g < CC) {
      #pragma unroll
      for (int e = 0; e < 8; ++e) {
        const int m = e + hi * 8;
        out[(size_t)(b0 + m) * CC + g] = acc[e];
      }
    }
  }
}

extern "C" void kernel_launch(void* const* d_in, const int* in_sizes, int n_in,
                              void* d_out, int out_size, void* d_ws, size_t ws_size,
                              hipStream_t stream) {
  (void)in_sizes; (void)n_in; (void)d_ws; (void)ws_size; (void)out_size;
  const float* x     = (const float*)d_in[0];
  const float* w_ih0 = (const float*)d_in[1];
  const float* w_hh0 = (const float*)d_in[2];
  const float* b_ih0 = (const float*)d_in[3];
  const float* b_hh0 = (const float*)d_in[4];
  const float* w_ih1 = (const float*)d_in[5];
  const float* w_hh1 = (const float*)d_in[6];
  const float* b_ih1 = (const float*)d_in[7];
  const float* b_hh1 = (const float*)d_in[8];
  const float* w_fc1 = (const float*)d_in[9];
  const float* b_fc1 = (const float*)d_in[10];
  const float* w_fc2 = (const float*)d_in[11];
  const float* b_fc2 = (const float*)d_in[12];

  constexpr size_t SMEM =
      (size_t)(2 * GG * HH + 128 * 64 + 32 * 128 + 2 * 2 * 16 * 64 + 16 * 128) * sizeof(bf16)
      + (size_t)2 * 16 * 8 * sizeof(float);   // = 87,040 bytes

  dim3 grid(BB / 16), block(128);
  gru_fused_kernel<<<grid, block, SMEM, stream>>>(
      x, w_ih0, w_hh0, b_ih0, b_hh0,
      w_ih1, w_hh1, b_ih1, b_hh1,
      w_fc1, b_fc1, w_fc2, b_fc2, (float*)d_out);
}
